// MultiHeadGATLayer_38714835206281
// MI455X (gfx1250) — matrix operand order repaired and checked by reading, not compiled, not run on hardware
//
#include <hip/hip_runtime.h>
#include <hip/hip_bf16.h>

#define ALPHA 0.2f

typedef __attribute__((ext_vector_type(16))) __bf16 v16bf;
typedef __attribute__((ext_vector_type(8)))  float  v8f;
typedef __attribute__((ext_vector_type(4)))  unsigned u32x4;
typedef __attribute__((ext_vector_type(8)))  int  i32x8;
typedef __attribute__((ext_vector_type(4)))  int  i32x4;

union ABPack { v16bf v; uint4 q[2]; unsigned u[8]; };

#if __has_builtin(__builtin_amdgcn_tensor_load_to_lds) && __has_builtin(__builtin_amdgcn_s_wait_tensorcnt)
#define HAVE_TDM 1
#else
#define HAVE_TDM 0
#endif

__device__ __forceinline__ unsigned bf16_bits(float f) {
    unsigned x = __float_as_uint(f);
    unsigned r = (x + 0x7FFFu + ((x >> 16) & 1u)) >> 16;   // RNE to bf16
    return r & 0xFFFFu;
}
__device__ __forceinline__ unsigned pack2(float f0, float f1) {
    return bf16_bits(f0) | (bf16_bits(f1) << 16);
}

__device__ __forceinline__ void atomicMaxF(float* addr, float val) {
    int* ai = (int*)addr;
    int old = __float_as_int(*addr);
    while (__int_as_float(old) < val) {
        int assumed = old;
        old = atomicCAS(ai, assumed, __float_as_int(val));
        if (old == assumed) break;
    }
}

#if HAVE_TDM
// 1-D TDM copy: ndw dwords from global gsrc -> LDS byte offset ldsOff.
// D# per CDNA5 ISA: group0 = {flags, lds_addr, gaddr},
// group1 = {data_size=4B, tensor_dim0=tile_dim0=ndw, tensor_dim1=1}.
__device__ __forceinline__ void tdm_copy_to_lds(const void* gsrc, unsigned ldsOff, unsigned ndw) {
    unsigned long long ga = (unsigned long long)gsrc;
    u32x4 g0;
    g0.x = 0x1u;                                            // count=1 (valid user desc)
    g0.y = ldsOff;                                          // lds_addr (bytes)
    g0.z = (unsigned)(ga & 0xFFFFFFFFu);                    // global_addr[31:0]
    g0.w = (unsigned)((ga >> 32) & 0x01FFFFFFu) | (2u << 30); // addr[56:32], type=2
    i32x8 g1 = (i32x8)0;
    g1[0] = (int)(2u << 16);                                // data_size = 4 bytes
    g1[1] = (int)((ndw & 0xFFFFu) << 16);                   // tensor_dim0[15:0]
    g1[2] = (int)((ndw >> 16) | (1u << 16));                // tensor_dim0[31:16], tensor_dim1=1
    g1[3] = (int)((ndw & 0xFFFFu) << 16);                   // tile_dim0 = ndw
    g1[5] = (int)ndw;                                       // tensor_dim0_stride (don't-care for 1 row)
    i32x4 z4 = (i32x4)0;
#if defined(__clang_major__) && __clang_major__ >= 23
    i32x8 z8 = (i32x8)0;
    __builtin_amdgcn_tensor_load_to_lds(g0, g1, z4, z4, z8, 0);
#else
    __builtin_amdgcn_tensor_load_to_lds(g0, g1, z4, z4, 0);
#endif
}
#endif

// ---------------------------------------------------------------------------
// Kernel 0: init output to zero, seg_max to -inf, seg_sum to zero
// ---------------------------------------------------------------------------
__global__ void init_k(float* __restrict__ out, float* __restrict__ seg_max,
                       float* __restrict__ seg_sum, int total_out, int hn) {
    int i = blockIdx.x * blockDim.x + threadIdx.x;
    if (i < total_out) out[i] = 0.0f;
    if (i < hn) { seg_max[i] = -__builtin_inff(); seg_sum[i] = 0.0f; }
}

// ---------------------------------------------------------------------------
// Kernel 1a: W (fp32, HxKxN row-major) -> Wbf packed-bf16 B fragments:
// layout [head][kk(16)][nt(8)][lane(32)][j(8)] ; dword j = B[kk*32+kb+(j<4 ? 2j : 16+2(j-4)) .. +1, nt*16+col]
// with col = lane&15, kb = (lane>>4)*8  (ISA 16-bit B-matrix layout)
// ---------------------------------------------------------------------------
__global__ void conv_w_k(const float* __restrict__ Wmat, unsigned* __restrict__ Wbf) {
    int idx = blockIdx.x * blockDim.x + threadIdx.x;   // 8*16*8*32*8 = 262144
    if (idx >= 262144) return;
    int j    = idx & 7;
    int lane = (idx >> 3) & 31;
    int nt   = (idx >> 8) & 7;
    int kk   = (idx >> 11) & 15;
    int head = idx >> 15;
    int col = nt * 16 + (lane & 15);
    int k   = kk * 32 + (lane >> 4) * 8 + ((j < 4) ? 2 * j : 16 + 2 * (j - 4));
    const float* Wq = Wmat + (size_t)head * 512 * 128;
    Wbf[idx] = pack2(Wq[(size_t)k * 128 + col], Wq[(size_t)(k + 1) * 128 + col]);
}

// ---------------------------------------------------------------------------
// Kernel 1b: h (fp32, NxK row-major) -> hbf packed-bf16 A fragments:
// layout [mt][kk(16)][lane(32)][j(8)] ; j<4 -> K=kbA+2j, j>=4 -> K=kbA+16+2(j-4)
// with row = mt*16 + (lane&15), kbA = (lane>>4)*8  (ISA 16-bit A-matrix layout)
// ---------------------------------------------------------------------------
__global__ void conv_h_k(const float* __restrict__ hmat, unsigned* __restrict__ hbf,
                         int N, int total) {
    int idx = blockIdx.x * blockDim.x + threadIdx.x;
    if (idx >= total) return;
    int j    = idx & 7;
    int lane = (idx >> 3) & 31;
    int kk   = (idx >> 8) & 15;
    int mt   = idx >> 12;
    int row = mt * 16 + (lane & 15);
    if (row >= N) row = N - 1;
    int k = kk * 32 + (lane >> 4) * 8 + ((j < 4) ? 2 * j : 16 + 2 * (j - 4));
    const float* hp = hmat + (size_t)row * 512 + k;
    hbf[idx] = pack2(hp[0], hp[1]);
}

// ---------------------------------------------------------------------------
// Kernel 2: Wh[h,n,:] = h[n,:] @ W[h,:,:] via v_wmma_f32_16x16x32_bf16.
// 256 thr = 8 waves; wave owns a 16-row strip; B fragments for the head are
// TDM-staged into LDS in two 64 KB phases and shared by all 8 waves.
// grid = (ceil(N/128), H)
// ---------------------------------------------------------------------------
__global__ void gemm_wh_k(const unsigned* __restrict__ hbf, const unsigned* __restrict__ Wbf,
                          float* __restrict__ Wh, int N) {
    __shared__ unsigned sB[16384];                 // 64 KB = 8 ksteps of B fragments

    const int head = blockIdx.y;
    const int wave = threadIdx.x >> 5;
    const int lane = threadIdx.x & 31;
    const int rowBase = blockIdx.x * 128 + wave * 16;
    const bool active = rowBase < N;
    const int mt = blockIdx.x * 8 + wave;

    v8f acc[8];
    for (int i = 0; i < 8; ++i)
        for (int j = 0; j < 8; ++j) acc[i][j] = 0.0f;

    const unsigned* WbfH = Wbf + (size_t)head * 32768;

    for (int ph = 0; ph < 2; ++ph) {
        const unsigned* gsrc = WbfH + (size_t)ph * 16384;
#if HAVE_TDM
        if (threadIdx.x == 0) {
            tdm_copy_to_lds(gsrc, 0u, 16384u);
            __builtin_amdgcn_s_wait_tensorcnt(0);
        }
        // The TDM descriptor carries the LDS address as plain data, so the
        // compiler sees no stores to sB and would fold the ds_loads to undef.
        // Escape sB into a memory-clobbering asm so its contents are treated
        // as externally written.
        asm volatile("" : : "r"(&sB[0]) : "memory");
#else
        for (int c = (int)threadIdx.x; c < 4096; c += 256)
            ((uint4*)sB)[c] = ((const uint4*)gsrc)[c];
#endif
        __syncthreads();

        if (active) {
            for (int kk2 = 0; kk2 < 8; ++kk2) {
                int kk = ph * 8 + kk2;
                ABPack a;
                const uint4* ap = (const uint4*)(hbf + (((size_t)mt * 16 + kk) * 32 + lane) * 8);
                a.q[0] = ap[0];
                a.q[1] = ap[1];
                const uint4* bp = (const uint4*)sB;
                #pragma unroll
                for (int nt = 0; nt < 8; ++nt) {
                    ABPack b;
                    int fi = ((kk2 * 8 + nt) * 32 + lane) * 2;
                    b.q[0] = bp[fi];
                    b.q[1] = bp[fi + 1];
                    acc[nt] = __builtin_amdgcn_wmma_f32_16x16x32_bf16(
                        false, a.v, false, b.v, (short)0, acc[nt], false, false);
                }
            }
        }
        __syncthreads();
    }

    if (active) {
        const int col = lane & 15;
        #pragma unroll
        for (int nt = 0; nt < 8; ++nt) {
            #pragma unroll
            for (int v = 0; v < 8; ++v) {
                int row = rowBase + v + 8 * (lane >> 4);
                if (row < N)
                    Wh[((size_t)head * N + row) * 128 + nt * 16 + col] = acc[nt][v];
            }
        }
    }
}

// ---------------------------------------------------------------------------
// Kernel 3: s_src[h,n] = Wh[h,n,:].a_src[h,:] ; s_dst likewise (tiny)
// ---------------------------------------------------------------------------
__global__ void score_k(const float* __restrict__ Wh, const float* __restrict__ a,
                        float* __restrict__ s_src, float* __restrict__ s_dst, int N) {
    int idx = blockIdx.x * blockDim.x + threadIdx.x;
    if (idx >= 8 * N) return;
    int head = idx / N, n = idx - head * N;
    const float* w  = Wh + ((size_t)head * N + n) * 128;
    const float* as = a + (size_t)head * 256;
    const float* ad = as + 128;
    float ss = 0.f, sd = 0.f;
    #pragma unroll 4
    for (int j = 0; j < 128; ++j) { float v = w[j]; ss += v * as[j]; sd += v * ad[j]; }
    s_src[idx] = ss; s_dst[idx] = sd;
}

// ---------------------------------------------------------------------------
// Kernel 4: segment max of e = leaky_relu(s_src[src]+s_dst[dst]) over src
// ---------------------------------------------------------------------------
__global__ void edge_max_k(const int* __restrict__ ei, const float* __restrict__ s_src,
                           const float* __restrict__ s_dst, float* __restrict__ seg_max,
                           int N, int E) {
    int e = blockIdx.x * blockDim.x + threadIdx.x;
    if (e >= E) return;
    int head = blockIdx.y;
    int s = ei[e], d = ei[E + e];
    float v = s_src[head * N + s] + s_dst[head * N + d];
    v = (v > 0.f) ? v : ALPHA * v;
    atomicMaxF(&seg_max[head * N + s], v);
}

// ---------------------------------------------------------------------------
// Kernel 5: segment sum of exp(e - seg_max[src])
// ---------------------------------------------------------------------------
__global__ void edge_sum_k(const int* __restrict__ ei, const float* __restrict__ s_src,
                           const float* __restrict__ s_dst, const float* __restrict__ seg_max,
                           float* __restrict__ seg_sum, int N, int E) {
    int e = blockIdx.x * blockDim.x + threadIdx.x;
    if (e >= E) return;
    int head = blockIdx.y;
    int s = ei[e], d = ei[E + e];
    float v = s_src[head * N + s] + s_dst[head * N + d];
    v = (v > 0.f) ? v : ALPHA * v;
    atomicAdd(&seg_sum[head * N + s], __expf(v - seg_max[head * N + s]));
}

// ---------------------------------------------------------------------------
// Kernel 6: out[src, head*128+j] += att * Wh[head, dst, j]
// one wave per (edge, head); lane handles a float4.
// ---------------------------------------------------------------------------
__global__ void scatter_k(const int* __restrict__ ei, const float* __restrict__ s_src,
                          const float* __restrict__ s_dst, const float* __restrict__ seg_max,
                          const float* __restrict__ seg_sum, const float* __restrict__ Wh,
                          float* __restrict__ out, int N, int E) {
    int head = blockIdx.y;
    int e = blockIdx.x * 8 + (threadIdx.x >> 5);
    if (e >= E) return;
    int lane = threadIdx.x & 31;
    int s = ei[e], d = ei[E + e];
    float v = s_src[head * N + s] + s_dst[head * N + d];
    v = (v > 0.f) ? v : ALPHA * v;
    float att = __expf(v - seg_max[head * N + s]) / seg_sum[head * N + s];

    const float4* whp = (const float4*)(Wh + ((size_t)head * N + d) * 128);
    float4 w4 = whp[lane];
    float* op = out + (size_t)s * 1024 + head * 128 + lane * 4;
    atomicAdd(op + 0, att * w4.x);
    atomicAdd(op + 1, att * w4.y);
    atomicAdd(op + 2, att * w4.z);
    atomicAdd(op + 3, att * w4.w);
}

// ---------------------------------------------------------------------------
extern "C" void kernel_launch(void* const* d_in, const int* in_sizes, int n_in,
                              void* d_out, int out_size, void* d_ws, size_t ws_size,
                              hipStream_t stream) {
    const float* h  = (const float*)d_in[0];
    const int*   ei = (const int*)d_in[1];
    const float* W  = (const float*)d_in[2];
    const float* a  = (const float*)d_in[3];
    float* out = (float*)d_out;

    const int N = in_sizes[0] / 512;     // 20000
    const int E = in_sizes[1] / 2;       // 640000
    const int HN = 8 * N;
    const int mtiles = (N + 15) / 16;    // 1250

    float* ws      = (float*)d_ws;
    float* Wh      = ws;                               // 8*N*128 floats
    float* s_src   = Wh + (size_t)8 * N * 128;
    float* s_dst   = s_src + HN;
    float* seg_max = s_dst + HN;
    float* seg_sum = seg_max + HN;
    unsigned* hbf  = (unsigned*)(seg_sum + HN);        // mtiles*16*32*8 dwords
    unsigned* Wbf  = hbf + (size_t)mtiles * 4096;      // 8*16*8*32*8 = 262144 dwords

    const int total_out = N * 1024;
    const int hbf_total = mtiles * 4096;

    init_k<<<dim3((total_out + 255) / 256), dim3(256), 0, stream>>>(
        out, seg_max, seg_sum, total_out, HN);

    conv_w_k<<<dim3(262144 / 256), dim3(256), 0, stream>>>(W, Wbf);
    conv_h_k<<<dim3((hbf_total + 255) / 256), dim3(256), 0, stream>>>(h, hbf, N, hbf_total);

    gemm_wh_k<<<dim3((N + 127) / 128, 8), dim3(256), 0, stream>>>(hbf, Wbf, Wh, N);

    score_k<<<dim3((HN + 255) / 256), dim3(256), 0, stream>>>(Wh, a, s_src, s_dst, N);

    edge_max_k<<<dim3((E + 255) / 256, 8), dim3(256), 0, stream>>>(
        ei, s_src, s_dst, seg_max, N, E);

    edge_sum_k<<<dim3((E + 255) / 256, 8), dim3(256), 0, stream>>>(
        ei, s_src, s_dst, seg_max, seg_sum, N, E);

    scatter_k<<<dim3((E + 7) / 8, 8), dim3(256), 0, stream>>>(
        ei, s_src, s_dst, seg_max, seg_sum, Wh, out, N, E);
}
